// MGSTGNN_52793738002629
// MI455X (gfx1250) — matrix operand order repaired and checked by reading, not compiled, not run on hardware
//
#include <hip/hip_runtime.h>

// ---------------------------------------------------------------------------
// MGSTGNN on MI455X (gfx1250): wave32, WMMA bf16 (f32 accumulate) for all
// matmuls, f32 VALU for LN / softmax statistics / pointwise GRU math.
// Round 1: cooperative P-tile sharing across waves (double-buffered LDS +
// split barriers), native bf16 converts, async global->LDS tile loads.
// ---------------------------------------------------------------------------

#define NN 2048   // nodes
#define BB 16     // batch
#define TT 12     // timesteps
#define DD 3      // input dim
#define EE 16     // embedding dim
#define HH 64     // hidden
#define CC 67     // D + H
#define CP80 80   // attention column pad (per batch) -> 5 tiles of 16
#define CP96 96   // node-matmul K pad -> 3 K-chunks of 32

typedef __attribute__((ext_vector_type(16))) __bf16         v16bf;
typedef __attribute__((ext_vector_type(16))) unsigned short v16us;
typedef __attribute__((ext_vector_type(8)))  unsigned short v8us;
typedef __attribute__((ext_vector_type(8)))  float          v8f;

union BFV { v16us u; v16bf b; };

__device__ __forceinline__ unsigned short f2bf(float f) {
  return __builtin_bit_cast(unsigned short, (__bf16)f);   // hw RNE convert
}

__device__ __forceinline__ v8f wmma_bf16(v16bf a, v16bf b, v8f c) {
  // D = A(16x32 bf16) * B(32x16 bf16) + C(16x16 f32)
  return __builtin_amdgcn_wmma_f32_16x16x32_bf16(false, a, false, b,
                                                 (short)0, c, false, false);
}

__device__ __forceinline__ v8f zero8() {
  v8f z; for (int i = 0; i < 8; ++i) z[i] = 0.f; return z;
}

// --------------------------------------------------------------------------
// ne = LayerNorm(node_emb + te)*w + b ; f32 copy + bf16 copy (16 cols)
// --------------------------------------------------------------------------
__global__ __launch_bounds__(256) void ln_ne_kernel(
    const float* __restrict__ node_emb, const float* __restrict__ te,
    const float* __restrict__ w, const float* __restrict__ bb,
    float* __restrict__ ne, unsigned short* __restrict__ neb) {
  int n = blockIdx.x * 256 + threadIdx.x;
  if (n >= NN) return;
  float x[EE]; float mu = 0.f;
  for (int e = 0; e < EE; ++e) { x[e] = node_emb[n*EE+e] + te[e]; mu += x[e]; }
  mu *= (1.f/EE);
  float var = 0.f;
  for (int e = 0; e < EE; ++e) { float d = x[e]-mu; var += d*d; }
  var *= (1.f/EE);
  float rs = rsqrtf(var + 1e-12f);
  for (int e = 0; e < EE; ++e) {
    float v = (x[e]-mu)*rs*w[e] + bb[e];
    ne[n*EE+e] = v;
    neb[n*EE+e] = f2bf(v);
  }
}

// --------------------------------------------------------------------------
// per-row softmax stats of s[n,m] = ne[n].ne[m]: row max and 1/sum(exp).
// ne tiles are staged through LDS with GLOBAL_LOAD_ASYNC_TO_LDS_B128
// (ASYNCcnt) -- each lane DMAs its own 64B row slice.
// --------------------------------------------------------------------------
__global__ __launch_bounds__(256) void softmax_stats_kernel(
    const float* __restrict__ ne, float* __restrict__ mx, float* __restrict__ inv) {
  __shared__ float tile[256*EE];
  int row = blockIdx.x * 256 + threadIdx.x;
  float my[EE];
  for (int e = 0; e < EE; ++e) my[e] = ne[row*EE+e];
  float m_run = -3.4e38f, s_run = 0.f;
  unsigned ldsdst = (unsigned)(unsigned long long)(&tile[threadIdx.x*EE]);
  for (int tb = 0; tb < NN; tb += 256) {
    __syncthreads();   // previous tile fully consumed
    unsigned voff = (unsigned)((tb + threadIdx.x)*EE*4);
    asm volatile(
      "global_load_async_to_lds_b128 %0, %1, %2 offset:0\n\t"
      "global_load_async_to_lds_b128 %0, %1, %2 offset:16\n\t"
      "global_load_async_to_lds_b128 %0, %1, %2 offset:32\n\t"
      "global_load_async_to_lds_b128 %0, %1, %2 offset:48"
      :: "v"(ldsdst), "v"(voff), "s"(ne) : "memory");
    asm volatile("s_wait_asynccnt 0" ::: "memory");
    __syncthreads();   // whole tile visible
    for (int mm = 0; mm < 256; ++mm) {
      float s = 0.f;
      for (int e = 0; e < EE; ++e) s += my[e]*tile[mm*EE+e];
      if (s > m_run) { s_run = s_run*__expf(m_run-s) + 1.f; m_run = s; }
      else           { s_run += __expf(s-m_run); }
    }
  }
  mx[row] = m_run;
  inv[row] = 1.f/s_run;
}

// --------------------------------------------------------------------------
// cat = [x, (z?)h]  -> bf16 copies: catb[b][m][96] and catbT[b*80+c][m]
// --------------------------------------------------------------------------
__global__ __launch_bounds__(256) void build_cat_kernel(
    const float* __restrict__ xsrc, int t,
    const float* __restrict__ h, const float* __restrict__ zr,
    unsigned short* __restrict__ catb, unsigned short* __restrict__ catbT) {
  int m = blockIdx.x * 256 + threadIdx.x;
  int b = blockIdx.y;
  float vals[CP80];
  const float* xp = xsrc + ((size_t)(b*TT + t)*NN + m)*DD;
  vals[0] = xp[0]; vals[1] = xp[1]; vals[2] = xp[2];
  const float* hp = h + ((size_t)b*NN + m)*HH;
  if (zr) {
    const float* zp = zr + ((size_t)b*NN + m)*(2*HH);
    for (int j = 0; j < HH; ++j) vals[DD+j] = hp[j]*zp[j];
  } else {
    for (int j = 0; j < HH; ++j) vals[DD+j] = hp[j];
  }
  for (int c = CC; c < CP80; ++c) vals[c] = 0.f;
  unsigned short* cb = catb + ((size_t)b*NN + m)*CP96;
  for (int c = 0; c < CP80; ++c) cb[c] = f2bf(vals[c]);
  for (int c = CP80; c < CP96; ++c) cb[c] = 0;
  for (int c = 0; c < CP80; ++c)
    catbT[((size_t)(b*CP80 + c))*NN + m] = f2bf(vals[c]);
}

// --------------------------------------------------------------------------
// Fused attention: xA = softmax(ne.neT) @ cat    (WMMA bf16, f32 accum)
// grid = (128 row-tiles, 2 col-halves) x 256 threads.
// Cooperative P: per round, wave w scores m-chunk (round*8+w), publishes the
// 16x32 bf16 P tile to double-buffered LDS; after one barrier all 8 waves
// consume all 8 P tiles against their 5 V column tiles (40 PV WMMAs/round).
// --------------------------------------------------------------------------
__global__ __launch_bounds__(256) void attn_kernel(
    const unsigned short* __restrict__ neb,
    const float* __restrict__ mx, const float* __restrict__ inv,
    const unsigned short* __restrict__ catbT,
    unsigned short* __restrict__ xAb) {
  const int tid  = threadIdx.x;
  const int lane = tid & 31, wave = tid >> 5;
  const int l15  = lane & 15, half = lane >> 4;
  const int Rbase = blockIdx.x * 16;
  __shared__ unsigned short pst[2][8][512];   // [buf][chunk-wave][16x32 bf16]

  float mxv[8], invv[8];
  for (int v = 0; v < 8; ++v) {
    int n = Rbase + v + 8*half;
    mxv[v] = mx[n]; invv[v] = inv[n];
  }
  // Score A operand: ne rows of this row tile (K=16 real, upper K half = 0)
  BFV aS;
  {
    v8us lo = *(const v8us*)(neb + (size_t)(Rbase + l15)*EE + half*8);
    for (int i = 0; i < 8; ++i) { aS.u[i] = lo[i]; aS.u[8+i] = 0; }
  }
  v8f acc[5];
  for (int tI = 0; tI < 5; ++tI) acc[tI] = zero8();
  const int tbase = (blockIdx.y*8 + wave)*5;   // col tiles [tbase..tbase+4] of 80

  for (int round = 0; round < 8; ++round) {
    const int buf = round & 1;
    const int Mb = (round*8 + wave)*32;        // this wave's chunk to score
    for (int hh = 0; hh < 2; ++hh) {
      BFV bS;
      {
        const unsigned short* np = neb + (size_t)(Mb + hh*16 + l15)*EE + half*8;
        v8us x0 = *(const v8us*)np;
        for (int i = 0; i < 8; ++i) { bS.u[i] = x0[i]; bS.u[8+i] = 0; }
      }
      v8f s = wmma_bf16(aS.b, bS.b, zero8());
      for (int v = 0; v < 8; ++v) {
        float p = __expf(s[v] - mxv[v]) * invv[v];
        pst[buf][wave][(v + 8*half)*32 + hh*16 + l15] = f2bf(p);
      }
    }
    __syncthreads();   // P tiles of this round published (double buffered)
    for (int cw = 0; cw < 8; ++cw) {
      const int Mc = (round*8 + cw)*32;
      BFV pA;          // reload P in 16x32 A-operand layout
      {
        const unsigned short* pp = &pst[buf][cw][l15*32 + half*16];
        v8us lo = *(const v8us*)pp;
        v8us hi = *(const v8us*)(pp + 8);
        for (int i = 0; i < 8; ++i) { pA.u[i] = lo[i]; pA.u[8+i] = hi[i]; }
      }
      for (int tI = 0; tI < 5; ++tI) {
        int col = (tbase + tI)*16 + l15;
        const unsigned short* vp = catbT + (size_t)col*NN + Mc + half*16;
        __builtin_prefetch((const void*)(vp + 256), 0, 0);
        BFV vB;
        v8us lo = *(const v8us*)vp;
        v8us hi = *(const v8us*)(vp + 8);
        for (int i = 0; i < 8; ++i) { vB.u[i] = lo[i]; vB.u[8+i] = hi[i]; }
        acc[tI] = wmma_bf16(pA.b, vB.b, acc[tI]);
      }
    }
  }
  // write xA (bf16, node_mm A-operand layout [b][n][96])
  for (int tI = 0; tI < 5; ++tI) {
    int col = (tbase + tI)*16 + l15;
    int b = col / CP80, c = col % CP80;
    for (int v = 0; v < 8; ++v) {
      int n = Rbase + v + 8*half;
      xAb[((size_t)b*NN + n)*CP96 + c] = f2bf(acc[tI][v]);
    }
  }
}

// --------------------------------------------------------------------------
// Per-node adaptive matmul: out = act( X@W0[n] + XA@W1[n] + ne[n]@bp )
// one block per node; W[n] built in LDS (E=16 dot per element), then
// wave-per-o-tile WMMAs with 3 K-chunks x 2 sources.
// --------------------------------------------------------------------------
__global__ void node_mm_kernel(
    const float* __restrict__ ne, const float* __restrict__ Wp,
    const float* __restrict__ bp,
    const unsigned short* __restrict__ catb,
    const unsigned short* __restrict__ xAb,
    float* __restrict__ outb, const int O, const int act) {
  __shared__ unsigned short WnB[2*128*CP96];   // [k][o][c] bf16
  __shared__ float biasS[128];
  __shared__ float neS[EE];
  const int n = blockIdx.x;
  const int tid = threadIdx.x;
  if (tid < EE) neS[tid] = ne[n*EE + tid];
  __syncthreads();

  const int total = 2*CP96*O;
  const int estride = 2*CC*O;                  // Wp e-stride
  for (int idx = tid; idx < total; idx += blockDim.x) {
    int o = idx % O;
    int c = (idx / O) % CP96;
    int k = idx / (O*CP96);
    float v = 0.f;
    if (c < CC) {
      const float* wp = Wp + (size_t)(k*CC + c)*O + o;
      __builtin_prefetch((const void*)(wp + estride), 0, 0);
      for (int e = 0; e < EE; ++e) v += neS[e]*wp[(size_t)e*estride];
    }
    WnB[(k*O + o)*CP96 + c] = f2bf(v);
  }
  for (int o = tid; o < O; o += blockDim.x) {
    float v = 0.f;
    for (int e = 0; e < EE; ++e) v += neS[e]*bp[e*O + o];
    biasS[o] = v;
  }
  __syncthreads();

  const int lane = tid & 31, wave = tid >> 5;
  const int l15 = lane & 15, half = lane >> 4;
  const int Ob = wave*16;
  v8f acc = zero8();
  for (int k = 0; k < 2; ++k) {
    const unsigned short* src = k ? xAb : catb;
    for (int kc = 0; kc < 3; ++kc) {
      BFV a;
      {
        const unsigned short* ap = src + ((size_t)l15*NN + n)*CP96 + kc*32 + half*8;
        v8us lo = *(const v8us*)ap;
        v8us hi = *(const v8us*)(ap + 16);
        for (int i = 0; i < 8; ++i) { a.u[i] = lo[i]; a.u[8+i] = hi[i]; }
      }
      BFV b;
      {
        const unsigned short* bv = &WnB[(k*O + Ob + l15)*CP96 + kc*32 + half*8];
        v8us lo = *(const v8us*)bv;
        v8us hi = *(const v8us*)(bv + 16);
        for (int i = 0; i < 8; ++i) { b.u[i] = lo[i]; b.u[8+i] = hi[i]; }
      }
      acc = wmma_bf16(a.b, b.b, acc);
    }
  }
  float bias = biasS[Ob + l15];
  for (int v = 0; v < 8; ++v) {
    float x = acc[v] + bias;
    x = act ? tanhf(x) : 1.f/(1.f + __expf(-x));
    int brow = v + 8*half;
    outb[((size_t)brow*NN + n)*O + Ob + l15] = x;
  }
}

// --------------------------------------------------------------------------
// GRU combine: h = r*h + (1-r)*hc ; optionally record last-2 timesteps
// --------------------------------------------------------------------------
__global__ __launch_bounds__(256) void combine_kernel(
    float* __restrict__ h, const float* __restrict__ zr,
    const float* __restrict__ hc, float* __restrict__ outslot) {
  int idx = blockIdx.x*256 + threadIdx.x;
  if (idx >= BB*NN*HH) return;
  int j = idx & (HH-1);
  int bn = idx >> 6;
  float r = zr[(size_t)bn*(2*HH) + HH + j];
  float hn = r*h[idx] + (1.f - r)*hc[idx];
  h[idx] = hn;
  if (outslot) {
    int b = bn >> 11, n = bn & (NN-1);
    outslot[(((size_t)b*2)*NN + n)*HH + j] = hn;
  }
}

// --------------------------------------------------------------------------
// skip head: cur_in = reshape(out0_last2 @ skip_W + skip_b) + source
// --------------------------------------------------------------------------
__global__ __launch_bounds__(256) void skip_kernel(
    const float* __restrict__ out0L2, const float* __restrict__ skip_W,
    const float* __restrict__ skip_b, const float* __restrict__ source,
    float* __restrict__ cur_in) {
  int idx = blockIdx.x*256 + threadIdx.x;
  if (idx >= BB*TT*NN*DD) return;
  int b = idx / (TT*NN*DD);
  int flat = idx % (TT*NN*DD);
  int o = flat / NN;         // 0..35
  int m = flat % NN;
  float acc = skip_b[o];
  for (int c2 = 0; c2 < 2; ++c2) {
    const float* cur = out0L2 + (((size_t)b*2 + c2)*NN + m)*HH;
    const float* w = skip_W + (o*2 + c2)*HH;
    for (int k = 0; k < HH; ++k) acc += cur[k]*w[k];
  }
  cur_in[idx] = acc + source[idx];
}

// --------------------------------------------------------------------------
// final: LN(concat(out0,out1) last-2) @ end_W + end_b -> (B,12,N,1)
// --------------------------------------------------------------------------
__global__ __launch_bounds__(256) void final_kernel(
    const float* __restrict__ out0L2, const float* __restrict__ out1L2,
    const float* __restrict__ nw, const float* __restrict__ nb,
    const float* __restrict__ end_W, const float* __restrict__ end_b,
    float* __restrict__ out) {
  int idx = blockIdx.x*256 + threadIdx.x;
  if (idx >= BB*NN) return;
  int b = idx >> 11, n = idx & (NN-1);
  float e[TT];
  for (int o = 0; o < TT; ++o) e[o] = end_b[o];
  for (int c2 = 0; c2 < 2; ++c2) {
    const float* p0 = out0L2 + (((size_t)b*2 + c2)*NN + n)*HH;
    const float* p1 = out1L2 + (((size_t)b*2 + c2)*NN + n)*HH;
    float mu = 0.f;
    for (int k = 0; k < HH; ++k) mu += p0[k] + p1[k];
    mu *= (1.f/(2*HH));
    float var = 0.f;
    for (int k = 0; k < HH; ++k) {
      float d0 = p0[k]-mu, d1 = p1[k]-mu;
      var += d0*d0 + d1*d1;
    }
    var *= (1.f/(2*HH));
    float rs = rsqrtf(var + 1e-12f);
    for (int k = 0; k < 2*HH; ++k) {
      float x = (k < HH) ? p0[k] : p1[k-HH];
      float v = (x-mu)*rs*nw[k] + nb[k];
      for (int o = 0; o < TT; ++o) e[o] += v*end_W[(o*2 + c2)*(2*HH) + k];
    }
  }
  for (int o = 0; o < TT; ++o) out[((size_t)b*TT + o)*NN + n] = e[o];
}

// --------------------------------------------------------------------------
extern "C" void kernel_launch(void* const* d_in, const int* in_sizes, int n_in,
                              void* d_out, int out_size, void* d_ws, size_t ws_size,
                              hipStream_t stream) {
  (void)in_sizes; (void)n_in; (void)out_size; (void)ws_size;
  const float* source   = (const float*)d_in[0];
  const float* node_emb = (const float*)d_in[1];
  const float* time_emb = (const float*)d_in[2];
  const float* gate_lnw = (const float*)d_in[3];
  const float* gate_lnb = (const float*)d_in[4];
  const float* gate_W   = (const float*)d_in[5];
  const float* gate_b   = (const float*)d_in[6];
  const float* upd_lnw  = (const float*)d_in[7];
  const float* upd_lnb  = (const float*)d_in[8];
  const float* upd_W    = (const float*)d_in[9];
  const float* upd_b    = (const float*)d_in[10];
  const float* skip_W   = (const float*)d_in[11];
  const float* skip_b   = (const float*)d_in[12];
  const float* norm_w   = (const float*)d_in[13];
  const float* norm_b   = (const float*)d_in[14];
  const float* end_W    = (const float*)d_in[15];
  const float* end_b    = (const float*)d_in[16];
  float* out = (float*)d_out;

  char* ws = (char*)d_ws;
  size_t off = 0;
  auto alloc = [&](size_t bytes) -> char* {
    char* p = ws + off;
    off += (bytes + 255) & ~(size_t)255;
    return p;
  };
  float* h0   = (float*)alloc((size_t)BB*NN*HH*4);
  float* hsA  = (float*)alloc((size_t)BB*NN*HH*4);
  float* hsB  = (float*)alloc((size_t)BB*NN*HH*4);
  float* zr   = (float*)alloc((size_t)BB*NN*2*HH*4);
  float* hc   = (float*)alloc((size_t)BB*NN*HH*4);
  float* neF  = (float*)alloc((size_t)NN*EE*4);
  float* mxv  = (float*)alloc((size_t)NN*4);
  float* invv = (float*)alloc((size_t)NN*4);
  float* out0L2 = (float*)alloc((size_t)BB*2*NN*HH*4);
  float* out1L2 = (float*)alloc((size_t)BB*2*NN*HH*4);
  float* cur_in = (float*)alloc((size_t)BB*TT*NN*DD*4);
  unsigned short* neb   = (unsigned short*)alloc((size_t)NN*EE*2);
  unsigned short* catb  = (unsigned short*)alloc((size_t)BB*NN*CP96*2);
  unsigned short* catbT = (unsigned short*)alloc((size_t)BB*CP80*NN*2);
  unsigned short* xAb   = (unsigned short*)alloc((size_t)BB*NN*CP96*2);

  hipMemsetAsync(h0,  0, (size_t)BB*NN*HH*4, stream);
  hipMemsetAsync(hsA, 0, (size_t)BB*NN*HH*4, stream);
  hipMemsetAsync(hsB, 0, (size_t)BB*NN*HH*4, stream);
  hipMemsetAsync(xAb, 0, (size_t)BB*NN*CP96*2, stream);   // keeps pad cols zero

  auto gru_step = [&](const float* xbase, int t, float* h, int cell, float* outslot) {
    // ---- gate GCN -> zr (sigmoid)
    ln_ne_kernel<<<8, 256, 0, stream>>>(node_emb, time_emb + t*EE,
        gate_lnw + cell*EE, gate_lnb + cell*EE, neF, neb);
    softmax_stats_kernel<<<8, 256, 0, stream>>>(neF, mxv, invv);
    build_cat_kernel<<<dim3(8,16), 256, 0, stream>>>(xbase, t, h, nullptr, catb, catbT);
    attn_kernel<<<dim3(128,2), 256, 0, stream>>>(neb, mxv, invv, catbT, xAb);
    node_mm_kernel<<<NN, 256, 0, stream>>>(neF,
        gate_W + (size_t)cell*EE*2*CC*(2*HH), gate_b + (size_t)cell*EE*(2*HH),
        catb, xAb, zr, 2*HH, 0);
    // ---- update GCN (cat2 = [x, z*h]) -> hc (tanh)
    ln_ne_kernel<<<8, 256, 0, stream>>>(node_emb, time_emb + t*EE,
        upd_lnw + cell*EE, upd_lnb + cell*EE, neF, neb);
    softmax_stats_kernel<<<8, 256, 0, stream>>>(neF, mxv, invv);
    build_cat_kernel<<<dim3(8,16), 256, 0, stream>>>(xbase, t, h, zr, catb, catbT);
    attn_kernel<<<dim3(128,2), 256, 0, stream>>>(neb, mxv, invv, catbT, xAb);
    node_mm_kernel<<<NN, 128, 0, stream>>>(neF,
        upd_W + (size_t)cell*EE*2*CC*HH, upd_b + (size_t)cell*EE*HH,
        catb, xAb, hc, HH, 1);
    // ---- h = r*h + (1-r)*hc
    combine_kernel<<<(BB*NN*HH)/256, 256, 0, stream>>>(h, zr, hc, outslot);
  };

  // layer 0 (cell 0)
  for (int t = 0; t < TT; ++t)
    gru_step(source, t, h0, 0, (t >= TT-2) ? out0L2 + (size_t)(t-(TT-2))*NN*HH : nullptr);

  // skip projection -> layer-1 input
  skip_kernel<<<(BB*TT*NN*DD + 255)/256, 256, 0, stream>>>(out0L2, skip_W, skip_b,
                                                           source, cur_in);

  // layer 1 (cells 1/2, alternating hidden states)
  for (int t = 0; t < TT; ++t) {
    int j = t & 1;
    gru_step(cur_in, t, j ? hsB : hsA, 1 + j,
             (t >= TT-2) ? out1L2 + (size_t)(t-(TT-2))*NN*HH : nullptr);
  }

  // final LN + output head
  final_kernel<<<(BB*NN + 255)/256, 256, 0, stream>>>(out0L2, out1L2, norm_w, norm_b,
                                                      end_W, end_b, out);
}
